// GroupedQueryAttention_71597104824907
// MI455X (gfx1250) — matrix operand order repaired and checked by reading, not compiled, not run on hardware
//
#include <hip/hip_runtime.h>

#define SEQ 2048
#define DIM 2048
#define NQH 32    // total query heads (8 groups * 4)
#define NKV 8     // kv heads (groups)
#define DQK 64    // head dim (== dv)

typedef __attribute__((ext_vector_type(16))) __bf16 v16bf;
typedef __attribute__((ext_vector_type(8)))  float  v8f;
typedef __attribute__((ext_vector_type(4)))  int    v4i;

// ---- CDNA5 async global->LDS path (ASYNCcnt tracked), with sync fallback ----
#if defined(__has_builtin)
#  if __has_builtin(__builtin_amdgcn_global_load_async_to_lds_b128)
#    define ASYNC_LDS 1
#  endif
#  if __has_builtin(__builtin_amdgcn_s_wait_asynccnt)
#    define HAVE_WAIT_ASYNC_BUILTIN 1
#  endif
#endif
#ifndef ASYNC_LDS
#  define ASYNC_LDS 0
#endif

typedef __attribute__((address_space(1))) v4i* gv4ip;   // global v4i*
typedef __attribute__((address_space(3))) v4i* lv4ip;   // LDS v4i*

__device__ __forceinline__ void stage_b128(void* lds_dst, const void* gsrc) {
#if ASYNC_LDS
    __builtin_amdgcn_global_load_async_to_lds_b128(
        (gv4ip)(unsigned long long)gsrc,
        (lv4ip)(unsigned)(unsigned long long)lds_dst,
        0, 0);
#else
    *(uint4*)lds_dst = *(const uint4*)gsrc;
#endif
}

__device__ __forceinline__ void wait_stage() {
#if ASYNC_LDS
#  if defined(HAVE_WAIT_ASYNC_BUILTIN)
    __builtin_amdgcn_s_wait_asynccnt(0);
#  else
    asm volatile("s_wait_asynccnt 0x0" ::: "memory");
#  endif
#endif
}

union BF16Frag {
    v16bf v;
    unsigned short u[16];
    uint4 q[2];
};

__device__ __forceinline__ unsigned short f2bf(float f) {
    unsigned int u = __float_as_uint(f);
    u += 0x7fffu + ((u >> 16) & 1u);   // round-to-nearest-even
    return (unsigned short)(u >> 16);
}

__device__ __forceinline__ void zero8(v8f& a) {
#pragma unroll
    for (int i = 0; i < 8; ++i) a[i] = 0.0f;
}

// A fragment (16x32 bf16, MxK): lane m=(lane&15) holds row m; kb=(lane>>4)*8;
// u[0..7] = K kb..kb+7, u[8..15] = K kb+16..kb+23  (ISA 7.12.2)
__device__ __forceinline__ v16bf load_a_frag(const unsigned short* base) {
    BF16Frag f;
    f.q[0] = *(const uint4*)(base);
    f.q[1] = *(const uint4*)(base + 16);
    return f.v;
}
// B fragment (32x16 bf16, KxN): lane n=(lane&15) holds col n; kb=(lane>>4)*8;
// u[0..7] = K kb..kb+7, u[8..15] = K kb+16..kb+23
__device__ __forceinline__ v16bf load_b_frag(const unsigned short* base) {
    BF16Frag f;
    f.q[0] = *(const uint4*)(base);
    f.q[1] = *(const uint4*)(base + 16);
    return f.v;
}

// ---------------------------------------------------------------- cvt f32->bf16 (x)
__global__ void __launch_bounds__(256) cvt_f32_bf16_kernel(const float* __restrict__ in,
                                                           unsigned short* __restrict__ out) {
    size_t i = ((size_t)blockIdx.x * blockDim.x + threadIdx.x) * 4;
    float4 f = *(const float4*)(in + i);
    ushort4 o;
    o.x = f2bf(f.x); o.y = f2bf(f.y); o.z = f2bf(f.z); o.w = f2bf(f.w);
    *(ushort4*)(out + i) = o;
}

// ---------------------------------------------------------------- transpose + cvt (weights)
// in:  f32 [.., k, n] row-major (ldin = n-extent)  -> out: bf16 [.., n, k] (ldout = k-extent)
__global__ void __launch_bounds__(256) transpose_cvt_kernel(const float* __restrict__ in,
                                                            unsigned short* __restrict__ out,
                                                            int ldin, int ldout,
                                                            long in_hs, long out_hs) {
    const int tid = threadIdx.x;
    const int k0  = blockIdx.x * 64;
    const int n0  = blockIdx.y * 64;
    in  += (size_t)blockIdx.z * in_hs;
    out += (size_t)blockIdx.z * out_hs;

    __shared__ __align__(16) unsigned short t[64 * 72];   // [n][k] stride 72

    {   // read [64 k][64 n] coalesced, convert, write transposed into LDS
        int r  = tid >> 2;          // k row 0..63
        int c0 = (tid & 3) * 16;    // n offset
        const float* src = in + (size_t)(k0 + r) * ldin + n0 + c0;
#pragma unroll
        for (int q4 = 0; q4 < 4; ++q4) {
            float4 a = *(const float4*)(src + q4 * 4);
            t[(c0 + q4 * 4 + 0) * 72 + r] = f2bf(a.x);
            t[(c0 + q4 * 4 + 1) * 72 + r] = f2bf(a.y);
            t[(c0 + q4 * 4 + 2) * 72 + r] = f2bf(a.z);
            t[(c0 + q4 * 4 + 3) * 72 + r] = f2bf(a.w);
        }
    }
    __syncthreads();
    {   // write [64 n][64 k] coalesced
        int n  = tid >> 2;
        int cc = (tid & 3) * 16;
        uint4* dst = (uint4*)(out + (size_t)(n0 + n) * ldout + k0 + cc);
        dst[0] = *(const uint4*)(t + n * 72 + cc);
        dst[1] = *(const uint4*)(t + n * 72 + cc + 8);
    }
}

// ---------------------------------------------------------------- projections
// MODE 0: Q (rope + 1/sqrt(dqk) folded in, out [H][SEQ][DQK])
// MODE 1: K (rope, out [H][SEQ][DQK])
// MODE 2: V (no rope, out transposed [H][DQK][SEQ])
template <int MODE>
__global__ void __launch_bounds__(256) proj_kernel(const unsigned short* __restrict__ xb, // [SEQ][DIM] bf16
                                                   const unsigned short* __restrict__ wT, // [H][DQK][DIM] bf16
                                                   unsigned short* __restrict__ out) {
    const int sblk = blockIdx.x;       // 16 blocks of 128 rows
    const int h    = blockIdx.y;
    const int tid  = threadIdx.x;
    const int wave = tid >> 5;
    const int lane = tid & 31;
    const int l16  = lane & 15;
    const int hi   = lane >> 4;

    constexpr int XT = 128 * 32, WT = 64 * 40;
    __shared__ __align__(16) unsigned short xt[2 * XT];   // [row][k] stride 32, double buffered
    __shared__ __align__(16) unsigned short wt[2 * WT];   // [n][k]   stride 40, double buffered

    const int row0 = sblk * 128;
    const unsigned short* wh = wT + (size_t)h * DQK * DIM;

    auto stage_tiles = [&](int kk, int b) {
        int r  = tid >> 1;
        int c0 = (tid & 1) * 16;
        const unsigned short* src = xb + (size_t)(row0 + r) * DIM + kk + c0;
        stage_b128(xt + b * XT + r * 32 + c0,     src);
        stage_b128(xt + b * XT + r * 32 + c0 + 8, src + 8);
        int n   = tid >> 2;
        int off = (tid & 3) * 8;
        stage_b128(wt + b * WT + n * 40 + off, wh + (size_t)n * DIM + kk + off);
    };

    v8f acc[4];
#pragma unroll
    for (int c = 0; c < 4; ++c) zero8(acc[c]);

    constexpr int NIT = DIM / 32;
    stage_tiles(0, 0);
    for (int it = 0; it < NIT; ++it) {
        wait_stage();
        __syncthreads();
        if (it + 1 < NIT) stage_tiles((it + 1) * 32, (it + 1) & 1);  // prefetch next tile
        const unsigned short* xb_ = xt + (it & 1) * XT;
        const unsigned short* wb_ = wt + (it & 1) * WT;

        v16bf af = load_a_frag(xb_ + (wave * 16 + l16) * 32 + hi * 8);
#pragma unroll
        for (int c = 0; c < 4; ++c) {
            v16bf bf = load_b_frag(wb_ + (c * 16 + l16) * 40 + hi * 8);
            acc[c] = __builtin_amdgcn_wmma_f32_16x16x32_bf16(false, af, false, bf,
                                                             (short)0, acc[c], false, false);
        }
        __syncthreads();
    }

    // Epilogue. C layout: acc[c][j] -> row = row0 + wave*16 + j + 8*hi, col = c*16 + l16
    if (MODE == 2) {
#pragma unroll
        for (int c = 0; c < 4; ++c)
#pragma unroll
            for (int j = 0; j < 8; ++j) {
                int row = row0 + wave * 16 + j + 8 * hi;
                int col = c * 16 + l16;
                out[((size_t)h * DQK + col) * SEQ + row] = f2bf(acc[c][j]);
            }
    } else {
        // RoPE: pair (2p, 2p+1) along head dim; partner element lives in lane^1
#pragma unroll
        for (int c = 0; c < 4; ++c) {
#pragma unroll
            for (int j = 0; j < 8; ++j) {
                float vv = acc[c][j];
                float pv = __shfl_xor(vv, 1, 32);
                int col  = c * 16 + l16;
                int row  = row0 + wave * 16 + j + 8 * hi;
                int p    = col >> 1;
                float invf = __expf(-(2.0f * (float)p / 64.0f) * 9.210340371976184f);
                float ang  = (float)row * invf;
                float sv, cv;
                __sincosf(ang, &sv, &cv);
                float r = (col & 1) ? (sv * pv + cv * vv) : (cv * vv - sv * pv);
                if (MODE == 0) r *= 0.125f;   // fold 1/sqrt(64) into Q
                out[((size_t)h * SEQ + row) * DQK + col] = f2bf(r);
            }
        }
    }
}

// ---------------------------------------------------------------- flash attention
// S^T = K * Q^T so each lane owns one query row: in-lane softmax + one shfl_xor(16),
// and P^T's C layout is exactly the A-fragment layout for P*V (register-only P).
__global__ void __launch_bounds__(256) attn_kernel(const unsigned short* __restrict__ qb,  // [NQH][SEQ][DQK]
                                                   const unsigned short* __restrict__ kbuf,// [NKV][SEQ][DQK]
                                                   const unsigned short* __restrict__ vtb, // [NKV][DQK][SEQ]
                                                   unsigned short* __restrict__ ob) {      // [SEQ][NQH*DQK]
    const int sblk = blockIdx.x;   // 16 query blocks of 128
    const int hq   = blockIdx.y;   // 32 query heads
    const int g    = hq >> 2;      // kv head
    const int tid  = threadIdx.x;
    const int wave = tid >> 5;
    const int lane = tid & 31;
    const int l16  = lane & 15;
    const int hi   = lane >> 4;

    constexpr int KT = 32 * 64, VT = 64 * 40;
    __shared__ __align__(16) unsigned short kt[2 * KT];   // [key][dqk] stride 64, double buffered
    __shared__ __align__(16) unsigned short vt[2 * VT];   // [v][key]   stride 40, double buffered

    const int rowb = sblk * 128 + wave * 16;   // this wave's 16 query rows
    const int qrow = rowb + l16;               // this lane's query row

    auto stage_tiles = [&](int key0, int b) {
        int r  = tid >> 3;
        int c0 = (tid & 7) * 8;
        stage_b128(kt + b * KT + r * 64 + c0, kbuf + ((size_t)g * SEQ + key0 + r) * DQK + c0);
        int n  = tid >> 2;
        int c1 = (tid & 3) * 8;
        stage_b128(vt + b * VT + n * 40 + c1, vtb + ((size_t)g * DQK + n) * SEQ + key0 + c1);
    };

    // Q^T B-fragments: lane n holds q-row qrow, K(dqk) = ks*32 + hi*8 + {0..7, 16..23}
    v16bf qf[2];
    {
        const unsigned short* base = qb + ((size_t)hq * SEQ + qrow) * DQK + hi * 8;
        qf[0] = load_b_frag(base);
        qf[1] = load_b_frag(base + 32);
    }

    v8f acc[4];
#pragma unroll
    for (int c = 0; c < 4; ++c) zero8(acc[c]);
    float m_run = -3e38f, l_run = 0.0f;

    const int nt = (sblk * 128 + 128) / 32;   // causal tile count for this block
    stage_tiles(0, 0);
    for (int it = 0; it < nt; ++it) {
        const int key0 = it * 32;
        wait_stage();
        __syncthreads();
        if (it + 1 < nt) stage_tiles(key0 + 32, (it + 1) & 1);   // prefetch next KV tile
        const unsigned short* kb_ = kt + (it & 1) * KT;
        const unsigned short* vb_ = vt + (it & 1) * VT;

        // wave-uniform skip: this wave's rows end at rowb+15; later tiles are fully masked
        if (key0 <= rowb + 15) {
            // S^T tiles: t -> keys key0+t*16 .. +15 (M), q rows (N)
            v8f s[2];
#pragma unroll
            for (int t = 0; t < 2; ++t) {
                v8f sc; zero8(sc);
#pragma unroll
                for (int ks = 0; ks < 2; ++ks) {
                    v16bf af = load_a_frag(kb_ + (t * 16 + l16) * 64 + ks * 32 + hi * 8);
                    sc = __builtin_amdgcn_wmma_f32_16x16x32_bf16(false, af, false, qf[ks],
                                                                 (short)0, sc, false, false);
                }
                s[t] = sc;
            }

            // causal masking only needed on diagonal tiles (wave-uniform test)
            if (key0 + 31 > rowb) {
#pragma unroll
                for (int t = 0; t < 2; ++t)
#pragma unroll
                    for (int j = 0; j < 8; ++j)
                        if (key0 + t * 16 + j + 8 * hi > qrow) s[t][j] = -3e38f;
            }

            // per-lane online softmax for query row `qrow`
            float mx = -3e38f;
#pragma unroll
            for (int t = 0; t < 2; ++t)
#pragma unroll
                for (int j = 0; j < 8; ++j) mx = fmaxf(mx, s[t][j]);
            mx = fmaxf(mx, __shfl_xor(mx, 16, 32));   // partner half-lane holds same row
            float m_new = fmaxf(m_run, mx);
            float alpha = __expf(m_run - m_new);
            m_run = m_new;

            BF16Frag pf;   // P A-fragment built fully in registers
            float sum = 0.0f;
#pragma unroll
            for (int t = 0; t < 2; ++t)
#pragma unroll
                for (int j = 0; j < 8; ++j) {
                    float p = __expf(s[t][j] - m_new);
                    sum += p;
                    pf.u[t * 8 + j] = f2bf(p);
                }
            sum += __shfl_xor(sum, 16, 32);
            l_run = l_run * alpha + sum;

            // rescale O: row of acc[c][j] is rowb + j + 8*hi; its alpha lives in lane j+8*hi
#pragma unroll
            for (int j = 0; j < 8; ++j) {
                float aj = __shfl(alpha, j + 8 * hi, 32);
#pragma unroll
                for (int c = 0; c < 4; ++c) acc[c][j] *= aj;
            }

#pragma unroll
            for (int c = 0; c < 4; ++c) {
                v16bf bf = load_b_frag(vb_ + (c * 16 + l16) * 40 + hi * 8);
                acc[c] = __builtin_amdgcn_wmma_f32_16x16x32_bf16(false, pf.v, false, bf,
                                                                 (short)0, acc[c], false, false);
            }
        }
        __syncthreads();
    }

    // epilogue: normalize and store O bf16 into concatenated layout [seq][hq*64 + v]
#pragma unroll
    for (int j = 0; j < 8; ++j) {
        float lr  = __shfl(l_run, j + 8 * hi, 32);
        float inv = 1.0f / lr;
        int row   = rowb + j + 8 * hi;
#pragma unroll
        for (int c = 0; c < 4; ++c)
            ob[(size_t)row * (NQH * DQK) + hq * DQK + c * 16 + l16] = f2bf(acc[c][j] * inv);
    }
}

// ---------------------------------------------------------------- output projection
// out[2048,2048] = Ocat[2048,2048](bf16) @ Wo (via pre-transposed woT bf16 [n][k])
__global__ void __launch_bounds__(256) out_gemm_kernel(const unsigned short* __restrict__ ob,
                                                       const unsigned short* __restrict__ woT,
                                                       float* __restrict__ out) {
    const int rb   = blockIdx.x;   // 128-row block
    const int cb   = blockIdx.y;   // 128-col block
    const int tid  = threadIdx.x;
    const int wave = tid >> 5;
    const int lane = tid & 31;
    const int l16  = lane & 15;
    const int hi   = lane >> 4;

    constexpr int AT = 128 * 32, WT = 128 * 40;
    __shared__ __align__(16) unsigned short at[2 * AT];   // [row][k] stride 32, double buffered
    __shared__ __align__(16) unsigned short wt[2 * WT];   // [n][k]   stride 40, double buffered

    auto stage_tiles = [&](int kk, int b) {
        int r  = tid >> 1;
        int c0 = (tid & 1) * 16;
        const unsigned short* asrc = ob + (size_t)(rb * 128 + r) * DIM + kk + c0;
        stage_b128(at + b * AT + r * 32 + c0,     asrc);
        stage_b128(at + b * AT + r * 32 + c0 + 8, asrc + 8);
        int n   = tid >> 1;
        int off = (tid & 1) * 16;
        const unsigned short* wsrc = woT + (size_t)(cb * 128 + n) * DIM + kk + off;
        stage_b128(wt + b * WT + n * 40 + off,     wsrc);
        stage_b128(wt + b * WT + n * 40 + off + 8, wsrc + 8);
    };

    v8f acc[8];
#pragma unroll
    for (int c = 0; c < 8; ++c) zero8(acc[c]);

    constexpr int NIT = DIM / 32;
    stage_tiles(0, 0);
    for (int it = 0; it < NIT; ++it) {
        wait_stage();
        __syncthreads();
        if (it + 1 < NIT) stage_tiles((it + 1) * 32, (it + 1) & 1);  // prefetch next tile
        const unsigned short* ab_ = at + (it & 1) * AT;
        const unsigned short* wb_ = wt + (it & 1) * WT;

        v16bf af = load_a_frag(ab_ + (wave * 16 + l16) * 32 + hi * 8);
#pragma unroll
        for (int c = 0; c < 8; ++c) {
            v16bf bf = load_b_frag(wb_ + (c * 16 + l16) * 40 + hi * 8);
            acc[c] = __builtin_amdgcn_wmma_f32_16x16x32_bf16(false, af, false, bf,
                                                             (short)0, acc[c], false, false);
        }
        __syncthreads();
    }

#pragma unroll
    for (int c = 0; c < 8; ++c)
#pragma unroll
        for (int j = 0; j < 8; ++j) {
            int row = rb * 128 + wave * 16 + j + 8 * hi;
            int col = cb * 128 + c * 16 + l16;
            out[(size_t)row * DIM + col] = acc[c][j];
        }
}

// ----------------------------------------------------------------
extern "C" void kernel_launch(void* const* d_in, const int* in_sizes, int n_in,
                              void* d_out, int out_size, void* d_ws, size_t ws_size,
                              hipStream_t stream) {
    (void)in_sizes; (void)n_in; (void)out_size; (void)ws_size;
    const float* x  = (const float*)d_in[0];
    const float* wq = (const float*)d_in[1];   // [8][4][2048][64] -> [32][2048][64]
    const float* wk = (const float*)d_in[2];   // [8][2048][64]
    const float* wv = (const float*)d_in[3];   // [8][2048][64]
    const float* wo = (const float*)d_in[4];   // [8][4][64][2048] -> [2048][2048]
    float* out = (float*)d_out;

    unsigned short* xb   = (unsigned short*)d_ws;                 //  8 MB  x bf16
    unsigned short* qbuf = xb   + (size_t)SEQ * DIM;              //  8 MB  q (rope+scale)
    unsigned short* kb   = qbuf + (size_t)NQH * SEQ * DQK;        //  2 MB  k (rope)
    unsigned short* vtb  = kb   + (size_t)NKV * SEQ * DQK;        //  2 MB  v^T
    unsigned short* obuf = vtb  + (size_t)NKV * DQK * SEQ;        //  8 MB  O concat
    unsigned short* wqT  = obuf + (size_t)SEQ * DIM;              //  8 MB  wq^T  [32][64][2048]
    unsigned short* wkT  = wqT  + (size_t)NQH * DQK * DIM;        //  2 MB  wk^T  [8][64][2048]
    unsigned short* wvT  = wkT  + (size_t)NKV * DQK * DIM;        //  2 MB  wv^T  [8][64][2048]
    unsigned short* woT  = wvT  + (size_t)NKV * DQK * DIM;        //  8 MB  wo^T  [2048][2048]

    cvt_f32_bf16_kernel<<<(SEQ * DIM) / 1024, 256, 0, stream>>>(x, xb);
    // weights: transpose + convert once (k-contiguous bf16 for async B staging)
    transpose_cvt_kernel<<<dim3(DIM / 64, 1, NQH), 256, 0, stream>>>(
        wq, wqT, DQK, DIM, (long)DIM * DQK, (long)DQK * DIM);
    transpose_cvt_kernel<<<dim3(DIM / 64, 1, NKV), 256, 0, stream>>>(
        wk, wkT, DQK, DIM, (long)DIM * DQK, (long)DQK * DIM);
    transpose_cvt_kernel<<<dim3(DIM / 64, 1, NKV), 256, 0, stream>>>(
        wv, wvT, DQK, DIM, (long)DIM * DQK, (long)DQK * DIM);
    transpose_cvt_kernel<<<dim3(DIM / 64, DIM / 64, 1), 256, 0, stream>>>(
        wo, woT, DIM, DIM, 0L, 0L);

    proj_kernel<0><<<dim3(SEQ / 128, NQH), 256, 0, stream>>>(xb, wqT, qbuf);
    proj_kernel<1><<<dim3(SEQ / 128, NKV), 256, 0, stream>>>(xb, wkT, kb);
    proj_kernel<2><<<dim3(SEQ / 128, NKV), 256, 0, stream>>>(xb, wvT, vtb);
    attn_kernel<<<dim3(SEQ / 128, NQH), 256, 0, stream>>>(qbuf, kb, vtb, obuf);
    out_gemm_kernel<<<dim3(SEQ / 128, DIM / 128), 256, 0, stream>>>(obuf, woT, out);
}
